// MolPNA_70300024701670
// MI455X (gfx1250) — matrix operand order repaired and checked by reading, not compiled, not run on hardware
//
#include <hip/hip_runtime.h>

// ---------------------------------------------------------------------------
// Types for gfx1250 WMMA (wave32): v_wmma_f32_16x16x32_bf16
// ---------------------------------------------------------------------------
typedef __bf16 bf16;
typedef __attribute__((ext_vector_type(16))) __bf16 v16bf;
typedef __attribute__((ext_vector_type(8)))  __bf16 v8bf;
typedef __attribute__((ext_vector_type(8)))  float  v8f;

#define NN      16384
#define NE      65536
#define NODE_IN 38
#define EDGE_IN 13
#define HID     256
#define NG      512
#define AVGLOG  1.3396215f   // sum(log(d+1)*hist)/sum(hist) from reference

// ---------------------------------------------------------------------------
// WMMA helpers
// ---------------------------------------------------------------------------
__device__ __forceinline__ v8f wmma_bf16(v16bf a, v16bf b, v8f c) {
  return __builtin_amdgcn_wmma_f32_16x16x32_bf16(false, a, false, b, (short)0, c,
                                                 false, false);
}

// A fragment (16x32 bf16) from row-major tile (row0,k0) with leading dim lda.
// lanes 0-15: row=lane, K {0..15}; lanes 16-31: row=lane-16, K {16..31}
__device__ __forceinline__ v16bf load_a(const bf16* p0, int lda) {
  const int lane = threadIdx.x & 31;
  const bf16* p = p0 + (lane & 15) * lda + ((lane >> 4) << 4);
  union { v16bf v; v8bf h[2]; } u;
  u.h[0] = *(const v8bf*)p;
  u.h[1] = *(const v8bf*)(p + 8);
  return u.v;
}

// B fragment (32x16 bf16) from packed tile: lane*16 contiguous bf16
__device__ __forceinline__ v16bf load_b(const bf16* tile) {
  const bf16* p = tile + (threadIdx.x & 31) * 16;
  union { v16bf v; v8bf h[2]; } u;
  u.h[0] = *(const v8bf*)p;
  u.h[1] = *(const v8bf*)(p + 8);
  return u.v;
}

// ---------------------------------------------------------------------------
// Float atomic max/min via ordered-int trick (buffers pre-filled with -/+inf)
// ---------------------------------------------------------------------------
__device__ __forceinline__ void atomicMaxF(float* a, float v) {
  if (v >= 0.0f) atomicMax((int*)a, __float_as_int(v));
  else           atomicMin((unsigned int*)a, __float_as_uint(v));
}
__device__ __forceinline__ void atomicMinF(float* a, float v) {
  if (v >= 0.0f) atomicMin((int*)a, __float_as_int(v));
  else           atomicMax((unsigned int*)a, __float_as_uint(v));
}

// ---------------------------------------------------------------------------
// Pack f32 row-major [K,N] weight into WMMA B-fragment tile order:
// dst[tile*512 + lane*16 + j] = src[(kt*32 + (lane>=16)*16 + j)*N + nt*16 + lane%16]
// ---------------------------------------------------------------------------
__global__ void pack_b_kernel(const float* __restrict__ src, bf16* __restrict__ dst,
                              int K, int N) {
  int idx = blockIdx.x * blockDim.x + threadIdx.x;
  if (idx >= K * N) return;
  int tile = idx >> 9;
  int r    = idx & 511;
  int lane = r >> 4, j = r & 15;
  int ntiles = N >> 4;
  int kt = tile / ntiles, nt = tile % ntiles;
  int f = kt * 32 + ((lane >> 4) << 4) + j;
  int c = nt * 16 + (lane & 15);
  dst[idx] = (bf16)src[(size_t)f * N + c];
}

__global__ void fill_u32_kernel(unsigned int* p, unsigned int v, size_t n) {
  size_t i = (size_t)blockIdx.x * blockDim.x + threadIdx.x;
  if (i < n) p[i] = v;
}

// ---------------------------------------------------------------------------
// Degree / node scalers
// ---------------------------------------------------------------------------
__global__ void degree_kernel(const int* __restrict__ ei, int* __restrict__ cnt) {
  int e = blockIdx.x * blockDim.x + threadIdx.x;
  if (e < NE) atomicAdd(&cnt[ei[NE + e]], 1);
}

__global__ void node_prep_kernel(const int* __restrict__ cnt, float* __restrict__ invc,
                                 float* __restrict__ hasnb, float* __restrict__ ampv,
                                 float* __restrict__ attv) {
  int n = blockIdx.x * blockDim.x + threadIdx.x;
  if (n >= NN) return;
  float c  = (float)cnt[n];
  float cc = fmaxf(c, 1.0f);
  float logd = logf(cc + 1.0f);
  invc[n]  = 1.0f / cc;
  hasnb[n] = (c > 0.0f) ? 1.0f : 0.0f;
  ampv[n]  = logd * (1.0f / AVGLOG);
  attv[n]  = AVGLOG / logd;
}

// ---------------------------------------------------------------------------
// Input projection h = x @ Wp + bp (K=38, tiny; VALU is fine)
// ---------------------------------------------------------------------------
__global__ void node_encode_kernel(const float* __restrict__ x, const float* __restrict__ Wp,
                                   const float* __restrict__ bp, float* __restrict__ h,
                                   bf16* __restrict__ hbf) {
  size_t i = (size_t)blockIdx.x * blockDim.x + threadIdx.x;
  if (i >= (size_t)NN * HID) return;
  int n = (int)(i >> 8), c = (int)(i & 255);
  float s = bp[c];
  #pragma unroll
  for (int k = 0; k < NODE_IN; ++k) s += x[(size_t)n * NODE_IN + k] * Wp[k * HID + c];
  h[i] = s;
  hbf[i] = (bf16)s;
}

// e = edge_attr @ We + be (K=13)
__global__ void edge_encode_kernel(const float* __restrict__ ea, const float* __restrict__ We,
                                   const float* __restrict__ be, bf16* __restrict__ ebf) {
  size_t i = (size_t)blockIdx.x * blockDim.x + threadIdx.x;
  if (i >= (size_t)NE * HID) return;
  int e = (int)(i >> 8), c = (int)(i & 255);
  float s = be[c];
  #pragma unroll
  for (int k = 0; k < EDGE_IN; ++k) s += ea[(size_t)e * EDGE_IN + k] * We[k * HID + c];
  ebf[i] = (bf16)s;
}

// ---------------------------------------------------------------------------
// Fused edge message GEMM + scatter reduction.
// 32 edges/block, 8 waves; wave w owns N-tiles {w, w+8, ...} of 64 (cols 0..1023).
// m = h[dst]@W0 + h[src]@W1 + e@W2 + bpre  -> atomics into sum/sq/max/min.
// ---------------------------------------------------------------------------
__global__ __launch_bounds__(256) void edge_msg_kernel(
    const int* __restrict__ ei, const bf16* __restrict__ hbf, const bf16* __restrict__ ebf,
    const bf16* __restrict__ pkpre,  // 12 packed [256,256] matrices, order (j*4+t)
    const float* __restrict__ bpre,  // [1024]
    float* __restrict__ sum, float* __restrict__ sq,
    float* __restrict__ mx, float* __restrict__ mn) {
  __shared__ __align__(16) bf16 smA[3][32][264];  // dst-rows, src-rows, e-rows
  __shared__ int smDst[32];
  const int tid = threadIdx.x;
  const int wave = tid >> 5, lane = tid & 31;
  const int ebase = blockIdx.x * 32;

  if (tid < 32) smDst[tid] = ei[NE + ebase + tid];
  for (int c = tid; c < 3 * 32 * 32; c += 256) {  // 16B chunks
    int k8 = c & 31;
    int r  = (c >> 5) & 31;
    int m  = c >> 10;
    int edge = ebase + r;
    const bf16* srcp;
    if (m == 0)      srcp = hbf + (size_t)ei[NE + edge] * HID + k8 * 8;
    else if (m == 1) srcp = hbf + (size_t)ei[edge] * HID + k8 * 8;
    else             srcp = ebf + (size_t)edge * HID + k8 * 8;
    *(v8bf*)&smA[m][r][k8 * 8] = *(const v8bf*)srcp;
  }
  __syncthreads();

  for (int nti = 0; nti < 8; ++nti) {
    const int nt = wave + nti * 8;          // 0..63 global col tile
    const int t = nt >> 4, nt4 = nt & 15;   // tower, tile within tower
    v8f acc0 = {}, acc1 = {};
    const bf16* b0 = pkpre + (size_t)(0 * 4 + t) * 65536;
    const bf16* b1 = pkpre + (size_t)(1 * 4 + t) * 65536;
    const bf16* b2 = pkpre + (size_t)(2 * 4 + t) * 65536;
    #pragma unroll
    for (int kt = 0; kt < 8; ++kt) {
      const int tileId = kt * 16 + nt4;
      v16bf Bd = load_b(b0 + (size_t)tileId * 512);
      v16bf Bs = load_b(b1 + (size_t)tileId * 512);
      v16bf Be = load_b(b2 + (size_t)tileId * 512);
      if (kt < 7) __builtin_prefetch(b0 + (size_t)(tileId + 16) * 512, 0, 0);
      const int k0 = kt * 32;
      acc0 = wmma_bf16(load_a(&smA[0][0][k0], 264), Bd, acc0);
      acc0 = wmma_bf16(load_a(&smA[1][0][k0], 264), Bs, acc0);
      acc0 = wmma_bf16(load_a(&smA[2][0][k0], 264), Be, acc0);
      acc1 = wmma_bf16(load_a(&smA[0][16][k0], 264), Bd, acc1);
      acc1 = wmma_bf16(load_a(&smA[1][16][k0], 264), Bs, acc1);
      acc1 = wmma_bf16(load_a(&smA[2][16][k0], 264), Be, acc1);
    }
    const int col = nt * 16 + (lane & 15);
    const float bias = bpre[col];
    const int rb = (lane >> 4) << 3;  // D layout: lanes 16-31 hold rows 8..15
    #pragma unroll
    for (int i = 0; i < 8; ++i) {
      float v0 = acc0[i] + bias;
      size_t i0 = (size_t)smDst[rb + i] * 1024 + col;
      atomicAdd(&sum[i0], v0); atomicAdd(&sq[i0], v0 * v0);
      atomicMaxF(&mx[i0], v0); atomicMinF(&mn[i0], v0);
      float v1 = acc1[i] + bias;
      size_t i1 = (size_t)smDst[16 + rb + i] * 1024 + col;
      atomicAdd(&sum[i1], v1); atomicAdd(&sq[i1], v1 * v1);
      atomicMaxF(&mx[i1], v1); atomicMinF(&mn[i1], v1);
    }
  }
}

// ---------------------------------------------------------------------------
// Post-NN: per 16-node tile, per tower: build aggr=[mean|max|min|std] (K=1024)
// in LDS, then acc_id (+ h@Wx), acc_amp, acc_att WMMA chains; combine with
// per-node amp/att scalers. Output post in bf16 [N,256].
// ---------------------------------------------------------------------------
__global__ __launch_bounds__(128) void post_kernel(
    const float* __restrict__ sum, const float* __restrict__ sq,
    const float* __restrict__ mx, const float* __restrict__ mn,
    const float* __restrict__ invc, const float* __restrict__ hasnb,
    const float* __restrict__ ampv, const float* __restrict__ attv,
    const bf16* __restrict__ hbf,
    const bf16* __restrict__ pkid, const bf16* __restrict__ pkamp,
    const bf16* __restrict__ pkatt, const bf16* __restrict__ pkx,
    const float* __restrict__ bpost, bf16* __restrict__ postbf) {
  __shared__ __align__(16) bf16 smAgg[16][1032];
  __shared__ __align__(16) bf16 smH[16][264];
  const int tid = threadIdx.x;
  const int wave = tid >> 5, lane = tid & 31;
  const int nbase = blockIdx.x * 16;

  for (int c = tid; c < 16 * 32; c += 128) {
    int r = c >> 5, k8 = c & 31;
    *(v8bf*)&smH[r][k8 * 8] = *(const v8bf*)(hbf + (size_t)(nbase + r) * HID + k8 * 8);
  }
  for (int t = 0; t < 4; ++t) {
    __syncthreads();
    for (int i = tid; i < 16 * 1024; i += 128) {
      int r = i >> 10, k = i & 1023;
      int s = k >> 8, g = k & 255;
      int node = nbase + r;
      size_t idx = (size_t)node * 1024 + t * 256 + g;
      float v;
      if (s == 0)      v = sum[idx] * invc[node];
      else if (s == 1) v = (hasnb[node] > 0.5f) ? mx[idx] : 0.0f;
      else if (s == 2) v = (hasnb[node] > 0.5f) ? mn[idx] : 0.0f;
      else {
        float mean = sum[idx] * invc[node];
        v = sqrtf(fmaxf(sq[idx] * invc[node] - mean * mean, 0.0f) + 1e-5f);
      }
      smAgg[r][k] = (bf16)v;
    }
    __syncthreads();
    v8f accI = {}, accA = {}, accT = {};
    const bf16* pid = pkid  + (size_t)t * 65536;
    const bf16* pam = pkamp + (size_t)t * 65536;
    const bf16* pat = pkatt + (size_t)t * 65536;
    const bf16* px  = pkx   + (size_t)t * 16384;
    #pragma unroll 4
    for (int kt = 0; kt < 32; ++kt) {
      const int tileId = kt * 4 + wave;
      v16bf A = load_a(&smAgg[0][kt * 32], 1032);
      accI = wmma_bf16(A, load_b(pid + (size_t)tileId * 512), accI);
      accA = wmma_bf16(A, load_b(pam + (size_t)tileId * 512), accA);
      accT = wmma_bf16(A, load_b(pat + (size_t)tileId * 512), accT);
    }
    #pragma unroll
    for (int kt = 0; kt < 8; ++kt) {
      const int tileId = kt * 4 + wave;
      accI = wmma_bf16(load_a(&smH[0][kt * 32], 264),
                       load_b(px + (size_t)tileId * 512), accI);
    }
    const int col = t * 64 + wave * 16 + (lane & 15);
    const float bias = bpost[col];
    const int rb = (lane >> 4) << 3;
    #pragma unroll
    for (int i = 0; i < 8; ++i) {
      int node = nbase + rb + i;
      float v = accI[i] + ampv[node] * accA[i] + attv[node] * accT[i] + bias;
      postbf[(size_t)node * HID + col] = (bf16)v;
    }
  }
}

// ---------------------------------------------------------------------------
// Generic [M,256] @ packed[256,256] + bias, optional ReLU (Wlin, Wr1)
// ---------------------------------------------------------------------------
__global__ __launch_bounds__(256) void gemm256_kernel(
    const bf16* __restrict__ A, const bf16* __restrict__ Bpk,
    const float* __restrict__ bias, float* __restrict__ C, int relu) {
  __shared__ __align__(16) bf16 smA[16][264];
  const int tid = threadIdx.x;
  const int wave = tid >> 5, lane = tid & 31;
  const int mbase = blockIdx.x * 16;
  for (int c = tid; c < 16 * 32; c += 256) {
    int r = c >> 5, k8 = c & 31;
    *(v8bf*)&smA[r][k8 * 8] = *(const v8bf*)(A + (size_t)(mbase + r) * HID + k8 * 8);
  }
  __syncthreads();
  for (int nti = 0; nti < 2; ++nti) {
    const int nt = wave * 2 + nti;
    v8f acc = {};
    #pragma unroll
    for (int kt = 0; kt < 8; ++kt) {
      const int tileId = kt * 16 + nt;
      if (kt < 7) __builtin_prefetch(Bpk + (size_t)(tileId + 16) * 512, 0, 0);
      acc = wmma_bf16(load_a(&smA[0][kt * 32], 264),
                      load_b(Bpk + (size_t)tileId * 512), acc);
    }
    const int col = nt * 16 + (lane & 15);
    const float b = bias[col];
    const int rb = (lane >> 4) << 3;
    #pragma unroll
    for (int i = 0; i < 8; ++i) {
      float v = acc[i] + b;
      if (relu) v = fmaxf(v, 0.0f);
      C[(size_t)(mbase + rb + i) * HID + col] = v;
    }
  }
}

// ---------------------------------------------------------------------------
// BatchNorm (two-pass) + ReLU
// ---------------------------------------------------------------------------
__global__ void bn_stats_kernel(const float* __restrict__ h, float* __restrict__ bsum,
                                float* __restrict__ bsq) {
  const int col = threadIdx.x;          // 256 channels
  const int rbase = blockIdx.x * 64;
  float s = 0.0f, q = 0.0f;
  for (int r = 0; r < 64; ++r) {
    float v = h[(size_t)(rbase + r) * HID + col];
    s += v; q += v * v;
  }
  atomicAdd(&bsum[col], s);
  atomicAdd(&bsq[col], q);
}

__global__ void bn_apply_kernel(float* __restrict__ h, bf16* __restrict__ hbf,
                                const float* __restrict__ bsum, const float* __restrict__ bsq,
                                const float* __restrict__ gamma, const float* __restrict__ beta) {
  size_t i = (size_t)blockIdx.x * blockDim.x + threadIdx.x;
  if (i >= (size_t)NN * HID) return;
  int c = (int)(i & 255);
  float mu  = bsum[c] * (1.0f / NN);
  float var = bsq[c] * (1.0f / NN) - mu * mu;
  float v = (h[i] - mu) * rsqrtf(var + 1e-5f) * gamma[c] + beta[c];
  v = fmaxf(v, 0.0f);
  h[i] = v;
  hbf[i] = (bf16)v;
}

// ---------------------------------------------------------------------------
// Global mean pool + readout head
// ---------------------------------------------------------------------------
__global__ void pool_scatter_kernel(const float* __restrict__ h, const int* __restrict__ batch,
                                    float* __restrict__ pool, float* __restrict__ gcnt) {
  size_t i = (size_t)blockIdx.x * blockDim.x + threadIdx.x;
  if (i >= (size_t)NN * HID) return;
  int n = (int)(i >> 8), c = (int)(i & 255);
  int g = batch[n];
  atomicAdd(&pool[(size_t)g * HID + c], h[i]);
  if (c == 0) atomicAdd(&gcnt[g], 1.0f);
}

__global__ void pool_finish_kernel(const float* __restrict__ pool, const float* __restrict__ gcnt,
                                   bf16* __restrict__ pbf) {
  size_t i = (size_t)blockIdx.x * blockDim.x + threadIdx.x;
  if (i >= (size_t)NG * HID) return;
  int g = (int)(i >> 8);
  pbf[i] = (bf16)(pool[i] / fmaxf(gcnt[g], 1.0f));
}

__global__ void head_kernel(const float* __restrict__ hid, const float* __restrict__ Wr2,
                            const float* __restrict__ br2, float* __restrict__ out) {
  int g = blockIdx.x * blockDim.x + threadIdx.x;
  if (g >= NG) return;
  float s = br2[0];
  for (int c = 0; c < HID; ++c) s += hid[(size_t)g * HID + c] * Wr2[c];
  out[g] = s;
}

// ---------------------------------------------------------------------------
// Host orchestration.  d_in order assumes JAX pytree flattening (sorted dict
// keys, lists in order):
//  0:x 1:edge_index 2:edge_attr 3:batch 4:Wp 5:Wr1 6:Wr2 7:bp 8:br1 9:br2
//  then per layer l (base=10+13l): +0 We, +1 Wlin, +2 Wpost_amp, +3 Wpost_att,
//  +4 Wpost_id, +5 Wpost_x, +6 Wpre, +7 be, +8 beta, +9 blin, +10 bpost,
//  +11 bpre, +12 gamma
// ---------------------------------------------------------------------------
extern "C" void kernel_launch(void* const* d_in, const int* in_sizes, int n_in,
                              void* d_out, int out_size, void* d_ws, size_t ws_size,
                              hipStream_t stream) {
  (void)in_sizes; (void)n_in; (void)out_size; (void)ws_size;
  const float* x     = (const float*)d_in[0];
  const int*   ei    = (const int*)d_in[1];
  const float* ea    = (const float*)d_in[2];
  const int*   batch = (const int*)d_in[3];
  const float* Wp    = (const float*)d_in[4];
  const float* Wr1   = (const float*)d_in[5];
  const float* Wr2   = (const float*)d_in[6];
  const float* bp    = (const float*)d_in[7];
  const float* br1   = (const float*)d_in[8];
  const float* br2   = (const float*)d_in[9];

  char* ws = (char*)d_ws;
  size_t off = 0;
  auto alloc = [&](size_t bytes) -> void* {
    void* p = ws + off;
    off = (off + bytes + 255) & ~(size_t)255;
    return p;
  };
  float* h     = (float*)alloc((size_t)NN * HID * 4);
  bf16* hbf    = (bf16*)alloc((size_t)NN * HID * 2);
  bf16* postbf = (bf16*)alloc((size_t)NN * HID * 2);
  bf16* ebf    = (bf16*)alloc((size_t)NE * HID * 2);
  float* sum   = (float*)alloc((size_t)NN * 1024 * 4);
  float* sq    = (float*)alloc((size_t)NN * 1024 * 4);
  float* mx    = (float*)alloc((size_t)NN * 1024 * 4);
  float* mn    = (float*)alloc((size_t)NN * 1024 * 4);
  int*   cnt   = (int*)alloc((size_t)NN * 4);
  float* invc  = (float*)alloc((size_t)NN * 4);
  float* hasnb = (float*)alloc((size_t)NN * 4);
  float* ampv  = (float*)alloc((size_t)NN * 4);
  float* attv  = (float*)alloc((size_t)NN * 4);
  float* bnsum = (float*)alloc(256 * 4);
  float* bnsq  = (float*)alloc(256 * 4);
  float* pool  = (float*)alloc((size_t)NG * HID * 4);
  float* gcnt  = (float*)alloc((size_t)NG * 4);
  bf16* poolbf = (bf16*)alloc((size_t)NG * HID * 2);
  float* hid   = (float*)alloc((size_t)NG * HID * 4);
  bf16* wr1pk  = (bf16*)alloc((size_t)65536 * 2);
  bf16 *pkpre[5], *pkid[5], *pkamp[5], *pkatt[5], *pkx[5], *pklin[5];
  for (int l = 0; l < 5; ++l) {
    pkpre[l] = (bf16*)alloc((size_t)12 * 65536 * 2);
    pkid[l]  = (bf16*)alloc((size_t)4 * 65536 * 2);
    pkamp[l] = (bf16*)alloc((size_t)4 * 65536 * 2);
    pkatt[l] = (bf16*)alloc((size_t)4 * 65536 * 2);
    pkx[l]   = (bf16*)alloc((size_t)4 * 16384 * 2);
    pklin[l] = (bf16*)alloc((size_t)65536 * 2);
  }

  // ---- weight packing (deterministic, every call) ----
  pack_b_kernel<<<256, 256, 0, stream>>>(Wr1, wr1pk, 256, 256);
  for (int l = 0; l < 5; ++l) {
    int base = 10 + 13 * l;
    const float* Wpre = (const float*)d_in[base + 6];
    for (int t = 0; t < 4; ++t)
      for (int j = 0; j < 3; ++j)
        pack_b_kernel<<<256, 256, 0, stream>>>(Wpre + (size_t)(t * 3 + j) * 65536,
                                               pkpre[l] + (size_t)(j * 4 + t) * 65536,
                                               256, 256);
    const float* Wid = (const float*)d_in[base + 4];
    const float* Wam = (const float*)d_in[base + 2];
    const float* Wat = (const float*)d_in[base + 3];
    const float* Wx  = (const float*)d_in[base + 5];
    for (int t = 0; t < 4; ++t) {
      pack_b_kernel<<<256, 256, 0, stream>>>(Wid + (size_t)t * 65536, pkid[l]  + (size_t)t * 65536, 1024, 64);
      pack_b_kernel<<<256, 256, 0, stream>>>(Wam + (size_t)t * 65536, pkamp[l] + (size_t)t * 65536, 1024, 64);
      pack_b_kernel<<<256, 256, 0, stream>>>(Wat + (size_t)t * 65536, pkatt[l] + (size_t)t * 65536, 1024, 64);
      pack_b_kernel<<<64,  256, 0, stream>>>(Wx  + (size_t)t * 16384, pkx[l]   + (size_t)t * 16384, 256, 64);
    }
    pack_b_kernel<<<256, 256, 0, stream>>>((const float*)d_in[base + 1], pklin[l], 256, 256);
  }

  // ---- degree stats & input projection ----
  hipMemsetAsync(cnt, 0, (size_t)NN * 4, stream);
  degree_kernel<<<NE / 256, 256, 0, stream>>>(ei, cnt);
  node_prep_kernel<<<NN / 256, 256, 0, stream>>>(cnt, invc, hasnb, ampv, attv);
  node_encode_kernel<<<NN * HID / 256, 256, 0, stream>>>(x, Wp, bp, h, hbf);

  // ---- layers ----
  for (int l = 0; l < 5; ++l) {
    int base = 10 + 13 * l;
    edge_encode_kernel<<<NE * HID / 256, 256, 0, stream>>>(
        ea, (const float*)d_in[base + 0], (const float*)d_in[base + 7], ebf);
    hipMemsetAsync(sum, 0, (size_t)NN * 1024 * 4, stream);
    hipMemsetAsync(sq,  0, (size_t)NN * 1024 * 4, stream);
    fill_u32_kernel<<<NN * 1024 / 256, 256, 0, stream>>>((unsigned int*)mx, 0xFF800000u, (size_t)NN * 1024);
    fill_u32_kernel<<<NN * 1024 / 256, 256, 0, stream>>>((unsigned int*)mn, 0x7F800000u, (size_t)NN * 1024);
    edge_msg_kernel<<<NE / 32, 256, 0, stream>>>(ei, hbf, ebf, pkpre[l],
                                                 (const float*)d_in[base + 11],
                                                 sum, sq, mx, mn);
    post_kernel<<<NN / 16, 128, 0, stream>>>(sum, sq, mx, mn, invc, hasnb, ampv, attv,
                                             hbf, pkid[l], pkamp[l], pkatt[l], pkx[l],
                                             (const float*)d_in[base + 10], postbf);
    gemm256_kernel<<<NN / 16, 256, 0, stream>>>(postbf, pklin[l],
                                                (const float*)d_in[base + 9], h, 0);
    hipMemsetAsync(bnsum, 0, 256 * 4, stream);
    hipMemsetAsync(bnsq,  0, 256 * 4, stream);
    bn_stats_kernel<<<NN / 64, 256, 0, stream>>>(h, bnsum, bnsq);
    bn_apply_kernel<<<NN * HID / 256, 256, 0, stream>>>(h, hbf, bnsum, bnsq,
                                                        (const float*)d_in[base + 12],
                                                        (const float*)d_in[base + 8]);
  }

  // ---- pooling + readout ----
  hipMemsetAsync(pool, 0, (size_t)NG * HID * 4, stream);
  hipMemsetAsync(gcnt, 0, (size_t)NG * 4, stream);
  pool_scatter_kernel<<<NN * HID / 256, 256, 0, stream>>>(h, batch, pool, gcnt);
  pool_finish_kernel<<<NG * HID / 256, 256, 0, stream>>>(pool, gcnt, poolbf);
  gemm256_kernel<<<NG / 16, 256, 0, stream>>>(poolbf, wr1pk, br1, hid, 1);
  head_kernel<<<(NG + 255) / 256, 256, 0, stream>>>(hid, Wr2, br2, (float*)d_out);
}
